// GAE_encoder_PNA_36429912605479
// MI455X (gfx1250) — matrix operand order repaired and checked by reading, not compiled, not run on hardware
//
#include <hip/hip_runtime.h>
#include <hip/hip_bf16.h>
#include <math.h>

// ---------------------------------------------------------------------------
// Types for WMMA fragments (wave32, v_wmma_f32_16x16x32_bf16)
// ---------------------------------------------------------------------------
typedef __bf16 v16bf __attribute__((ext_vector_type(16)));
typedef float  v8f   __attribute__((ext_vector_type(8)));
typedef unsigned uv4 __attribute__((ext_vector_type(4)));

union FragBF { uv4 u[2]; v16bf v; };

// A-matrix 16x32 bf16 fragment (contiguous k order, same as load_b_frag):
//  lane<16 : elements 0..15 = row[K+0..15]
//  lane>=16: elements 0..15 = row[K+16..31]
// row points at (base + K); half = lane>>4. 16B-aligned loads.
__device__ inline v16bf load_a_frag(const unsigned short* __restrict__ row, int half) {
  FragBF f;
  f.u[0] = *reinterpret_cast<const uv4*>(row + half * 16);
  f.u[1] = *reinterpret_cast<const uv4*>(row + half * 16 + 8);
  return f.v;
}

// B-matrix 32x16 bf16 fragment from a TRANSPOSED weight array WT[n][K]:
//  lane l holds column n=l&15, K-range (l<16 ? kb..kb+15 : kb+16..kb+31),
//  contiguous in WT. colptr points at WT + n*Kdim + kb + (lane>>4)*16.
__device__ inline v16bf load_b_frag(const unsigned short* __restrict__ colptr) {
  FragBF f;
  f.u[0] = *reinterpret_cast<const uv4*>(colptr);
  f.u[1] = *reinterpret_cast<const uv4*>(colptr + 8);
  return f.v;
}

__device__ inline v8f wmma_bf16(v16bf a, v16bf b, v8f c) {
  return __builtin_amdgcn_wmma_f32_16x16x32_bf16(false, a, false, b, (short)0, c,
                                                 false, false);
}

// ---------------------------------------------------------------------------
// Scalar helpers
// ---------------------------------------------------------------------------
__device__ inline unsigned short f2bf(float f) {  // round-to-nearest-even
  unsigned u = __float_as_uint(f);
  unsigned r = (u + 0x7FFFu + ((u >> 16) & 1u)) >> 16;
  return (unsigned short)r;
}
__device__ inline float bf2f(unsigned short h) {
  return __uint_as_float(((unsigned)h) << 16);
}
// order-preserving float<->uint encoding for atomicMax on floats
__device__ inline unsigned encf(float f) {
  int i = __float_as_int(f);
  unsigned u = (unsigned)i;
  return (i >= 0) ? (u | 0x80000000u) : ~u;
}
__device__ inline float decf(unsigned u) {
  unsigned v = (u & 0x80000000u) ? (u & 0x7FFFFFFFu) : ~u;
  return __int_as_float((int)v);
}
#define ENC_NEG_INF 0x007FFFFFu  // encf(-inf)
#define INV_AVG_DEG_LOG 1.2586314f  // 1 / (sum(log(1..4))/4) = 1/0.79451346
#define BN_EPS 1e-5f

// ---------------------------------------------------------------------------
// Column statistics (biased batch stats): one block per column
// ---------------------------------------------------------------------------
__global__ void col_stats_kernel(const float* __restrict__ X, int N, int C,
                                 float* __restrict__ MU, float* __restrict__ ISTD) {
  __shared__ float ss[256], sq[256];
  const int c = blockIdx.x, t = threadIdx.x;
  float s = 0.f, q = 0.f;
  for (int r = t; r < N; r += 256) {
    float v = X[(size_t)r * C + c];
    s += v; q += v * v;
  }
  ss[t] = s; sq[t] = q;
  __syncthreads();
  for (int ofs = 128; ofs > 0; ofs >>= 1) {
    if (t < ofs) { ss[t] += ss[t + ofs]; sq[t] += sq[t + ofs]; }
    __syncthreads();
  }
  if (t == 0) {
    float mu = ss[0] / (float)N;
    float var = sq[0] / (float)N - mu * mu;
    MU[c] = mu;
    ISTD[c] = rsqrtf(var + BN_EPS);
  }
}

// normalize input and cast to bf16; one block per row (256 cols)
__global__ void normalize_cast_kernel(const float* __restrict__ x,
                                      const float* __restrict__ MU,
                                      const float* __restrict__ ISTD,
                                      const float* __restrict__ g,
                                      const float* __restrict__ bt,
                                      unsigned short* __restrict__ XN) {
  int idx = blockIdx.x * 256 + threadIdx.x;
  int c = threadIdx.x;
  XN[idx] = f2bf((x[idx] - MU[c]) * ISTD[c] * g[c] + bt[c]);
}

// ---------------------------------------------------------------------------
// Weight prep (per branch)
// ---------------------------------------------------------------------------
// pre_W [512,256] fp32 -> WT [256][512] bf16 (transposed)
__global__ void transpose_preW_kernel(const float* __restrict__ preW,
                                      unsigned short* __restrict__ WT) {
  int idx = blockIdx.x * 256 + threadIdx.x;  // 256*512 ids
  int n = idx >> 9, k = idx & 511;
  WT[idx] = f2bf(preW[(size_t)k * 256 + n]);
}

// Fold post_W (dup-mean + 3x amp copies) through lin_W into
//   WQT [128][1024] (xn | raw stats path)   WPT [128][768] (amp-scaled path)
__global__ void comb_weights_kernel(const float* __restrict__ postW,
                                    const float* __restrict__ linW,
                                    unsigned short* __restrict__ WQT,
                                    unsigned short* __restrict__ WPT) {
  int idx = blockIdx.x * 256 + threadIdx.x;  // 1792*128 ids
  int r = idx >> 7, c = idx & 127;
  if (r >= 1792) return;
  int rows[6]; int nrows = 0;
  if (r < 1024) {
    int seg = r >> 8, rr = r & 255;
    if (seg == 0)      { rows[nrows++] = rr; }                       // x
    else if (seg == 1) { rows[nrows++] = 3328 + rr;                  // raw mean (x2)
                         rows[nrows++] = 3584 + rr; }
    else if (seg == 2) { rows[nrows++] = 3840 + rr; }                // raw std
    else               { rows[nrows++] = 4096 + rr; }                // raw max
  } else {
    int rp = r - 1024, seg = rp >> 8, rr = rp & 255;
    for (int k = 0; k < 3; ++k) {
      int base = 256 + 1024 * k;
      if (seg == 0)      { rows[nrows++] = base + rr;                // amp mean (x2)
                           rows[nrows++] = base + 256 + rr; }
      else if (seg == 1) { rows[nrows++] = base + 512 + rr; }        // amp std
      else               { rows[nrows++] = base + 768 + rr; }        // amp max
    }
  }
  float acc = 0.f;
  for (int j = 0; j < 128; ++j) {
    float w = 0.f;
    for (int t = 0; t < nrows; ++t) w += postW[(size_t)rows[t] * 128 + j];
    acc += w * linW[(size_t)j * 128 + c];
  }
  if (r < 1024) WQT[(size_t)c * 1024 + r] = f2bf(acc);
  else          WPT[(size_t)c * 768 + (r - 1024)] = f2bf(acc);
}

// b_comb = post_b @ lin_W + lin_b
__global__ void bcomb_kernel(const float* __restrict__ postB,
                             const float* __restrict__ linW,
                             const float* __restrict__ linB,
                             float* __restrict__ BCOMB) {
  int c = threadIdx.x;
  float acc = linB[c];
  for (int j = 0; j < 128; ++j) acc += postB[j] * linW[(size_t)j * 128 + c];
  BCOMB[c] = acc;
}

// ---------------------------------------------------------------------------
// Aggregation buffers
// ---------------------------------------------------------------------------
__global__ void init_agg_kernel(float* __restrict__ SUM, float* __restrict__ SSQ,
                                unsigned* __restrict__ MAXE, float* __restrict__ DEG,
                                int N) {
  int idx = blockIdx.x * 256 + threadIdx.x;  // grid == N blocks -> N*256 ids
  SUM[idx] = 0.f; SSQ[idx] = 0.f; MAXE[idx] = ENC_NEG_INF;
  if (idx < N) DEG[idx] = 0.f;
}

__global__ void deg_kernel(const int* __restrict__ ei, int E, float* __restrict__ DEG) {
  int e = blockIdx.x * 256 + threadIdx.x;
  if (e < E) atomicAdd(&DEG[ei[E + e]], 1.0f);
}

// ---------------------------------------------------------------------------
// Edge GEMM: m[e,:] = [xn[dst]|xn[src]] @ pre_W + pre_b, scatter to sum/sumsq/max.
// Block = 128 thr (4 waves), tile = 32 edges x 256 cols; wave owns 2 M x 4 N tiles.
// ---------------------------------------------------------------------------
__global__ void __launch_bounds__(128)
edge_gemm_kernel(const unsigned short* __restrict__ XN,
                 const unsigned short* __restrict__ WT,  // [256][512] bf16
                 const float* __restrict__ preB,
                 const int* __restrict__ ei, int E,
                 float* __restrict__ SUM, float* __restrict__ SSQ,
                 unsigned* __restrict__ MAXE) {
  __shared__ int ssrc[32], sdst[32];
  const int ebase = blockIdx.x * 32;
  const int tid = threadIdx.x;
  if (tid < 32) { ssrc[tid] = ei[ebase + tid]; sdst[tid] = ei[E + ebase + tid]; }
  __syncthreads();
  const int wave = tid >> 5, lane = tid & 31;
  const int l15 = lane & 15, half = lane >> 4;

  v8f acc[2][4];
  for (int j = 0; j < 4; ++j) {
    float pb = preB[wave * 64 + j * 16 + l15];
    for (int i = 0; i < 2; ++i)
      for (int r = 0; r < 8; ++r) acc[i][j][r] = pb;
  }

  for (int kb = 0; kb < 512; kb += 32) {
    const int* idx = (kb < 256) ? sdst : ssrc;  // h = [x_i, x_j] = [x_dst, x_src]
    const int koff = kb & 255;
    v16bf a[2];
    for (int i = 0; i < 2; ++i) {
      int node = idx[i * 16 + l15];
      a[i] = load_a_frag(XN + (size_t)node * 256 + koff, half);
    }
    for (int j = 0; j < 4; ++j) {
      int n = wave * 64 + j * 16 + l15;
      const unsigned short* bp = WT + (size_t)n * 512 + kb + half * 16;
      v16bf bfrag = load_b_frag(bp);
      if (kb + 32 < 512) __builtin_prefetch(bp + 32, 0, 0);
      for (int i = 0; i < 2; ++i) acc[i][j] = wmma_bf16(a[i], bfrag, acc[i][j]);
    }
  }

  // epilogue: C layout -> lane holds col l15, rows half*8 + r
  for (int i = 0; i < 2; ++i) {
    for (int r = 0; r < 8; ++r) {
      int mrow = i * 16 + half * 8 + r;
      int d = sdst[mrow];
      size_t base = (size_t)d * 256;
      for (int j = 0; j < 4; ++j) {
        int col = wave * 64 + j * 16 + l15;
        float v = acc[i][j][r];
        atomicAdd(&SUM[base + col], v);
        atomicAdd(&SSQ[base + col], v * v);
        atomicMax(&MAXE[base + col], encf(v));
      }
    }
  }
}

// scalar fallback for edge tail (E % 32)
__global__ void edge_tail_kernel(const unsigned short* __restrict__ XN,
                                 const unsigned short* __restrict__ WT,
                                 const float* __restrict__ preB,
                                 const int* __restrict__ ei, int E, int estart,
                                 float* __restrict__ SUM, float* __restrict__ SSQ,
                                 unsigned* __restrict__ MAXE) {
  int idx = blockIdx.x * 256 + threadIdx.x;
  int ne = E - estart;
  if (idx >= ne * 256) return;
  int e = estart + idx / 256, c = idx & 255;
  int s = ei[e], d = ei[E + e];
  float acc = preB[c];
  for (int k = 0; k < 256; ++k) {
    acc += bf2f(XN[(size_t)d * 256 + k]) * bf2f(WT[(size_t)c * 512 + k]);
    acc += bf2f(XN[(size_t)s * 256 + k]) * bf2f(WT[(size_t)c * 512 + 256 + k]);
  }
  size_t o = (size_t)d * 256 + c;
  atomicAdd(&SUM[o], acc);
  atomicAdd(&SSQ[o], acc * acc);
  atomicMax(&MAXE[o], encf(acc));
}

// per-node mean/std/max/amp -> bf16 stats [N][768] = [mean|std|max]
__global__ void finalize_kernel(const float* __restrict__ SUM,
                                const float* __restrict__ SSQ,
                                const unsigned* __restrict__ MAXE,
                                const float* __restrict__ DEG,
                                unsigned short* __restrict__ STATS,
                                float* __restrict__ AMP) {
  int i = blockIdx.x, c = threadIdx.x;
  float deg = DEG[i];
  float degc = fmaxf(deg, 1.f);
  float inv = 1.f / degc;
  int base = i * 256 + c;
  float mean = SUM[base] * inv;
  float msq = SSQ[base] * inv;
  float sd = sqrtf(fmaxf(msq - mean * mean, 0.f) + 1e-5f);
  float mx = (deg > 0.f) ? decf(MAXE[base]) : 0.f;
  int sb = i * 768 + c;
  STATS[sb] = f2bf(mean);
  STATS[sb + 256] = f2bf(sd);
  STATS[sb + 512] = f2bf(mx);
  if (c == 0) AMP[i] = logf(degc + 1.f) * INV_AVG_DEG_LOG;
}

// ---------------------------------------------------------------------------
// Node GEMM: o = [xn|stats]@Wq + amp*(stats@Wp) + b_comb.
// Block = 128 thr (4 waves), tile = 16 nodes x 128 cols; wave owns 2 N tiles,
// dual accumulators (Q always, P for K>=256).
// ---------------------------------------------------------------------------
__global__ void __launch_bounds__(128)
node_gemm_kernel(const unsigned short* __restrict__ XN,
                 const unsigned short* __restrict__ STATS,
                 const unsigned short* __restrict__ WQT,  // [128][1024]
                 const unsigned short* __restrict__ WPT,  // [128][768]
                 const float* __restrict__ BCOMB,
                 const float* __restrict__ AMP,
                 float* __restrict__ OUT) {
  const int nbase = blockIdx.x * 16;
  const int tid = threadIdx.x, wave = tid >> 5, lane = tid & 31;
  const int l15 = lane & 15, half = lane >> 4;

  v8f q[2], p[2];
  for (int t = 0; t < 2; ++t) {
    float bc = BCOMB[(wave * 2 + t) * 16 + l15];
    for (int r = 0; r < 8; ++r) { q[t][r] = bc; p[t][r] = 0.f; }
  }
  const int node = nbase + l15;
  for (int kb = 0; kb < 1024; kb += 32) {
    v16bf a;
    if (kb < 256) a = load_a_frag(XN + (size_t)node * 256 + kb, half);
    else          a = load_a_frag(STATS + (size_t)node * 768 + (kb - 256), half);
    for (int t = 0; t < 2; ++t) {
      int col = (wave * 2 + t) * 16 + l15;
      v16bf bq = load_b_frag(WQT + (size_t)col * 1024 + kb + half * 16);
      q[t] = wmma_bf16(a, bq, q[t]);
    }
    if (kb >= 256) {
      for (int t = 0; t < 2; ++t) {
        int col = (wave * 2 + t) * 16 + l15;
        v16bf bp = load_b_frag(WPT + (size_t)col * 768 + (kb - 256) + half * 16);
        p[t] = wmma_bf16(a, bp, p[t]);
      }
    }
  }
  for (int r = 0; r < 8; ++r) {
    int nr = nbase + half * 8 + r;
    float amp = AMP[nr];
    for (int t = 0; t < 2; ++t) {
      int col = (wave * 2 + t) * 16 + l15;
      OUT[(size_t)nr * 128 + col] = q[t][r] + amp * p[t][r];
    }
  }
}

// scalar fallback for node tail (N % 16)
__global__ void node_tail_kernel(const unsigned short* __restrict__ XN,
                                 const unsigned short* __restrict__ STATS,
                                 const unsigned short* __restrict__ WQT,
                                 const unsigned short* __restrict__ WPT,
                                 const float* __restrict__ BCOMB,
                                 const float* __restrict__ AMP,
                                 float* __restrict__ OUT, int nstart, int N) {
  int idx = blockIdx.x * 256 + threadIdx.x;
  int nn = N - nstart;
  if (idx >= nn * 128) return;
  int node = nstart + idx / 128, c = idx & 127;
  float q = BCOMB[c], p = 0.f;
  for (int k = 0; k < 256; ++k)
    q += bf2f(XN[(size_t)node * 256 + k]) * bf2f(WQT[(size_t)c * 1024 + k]);
  for (int k = 0; k < 768; ++k) {
    float sv = bf2f(STATS[(size_t)node * 768 + k]);
    q += sv * bf2f(WQT[(size_t)c * 1024 + 256 + k]);
    p += sv * bf2f(WPT[(size_t)c * 768 + k]);
  }
  OUT[(size_t)node * 128 + c] = q + AMP[node] * p;
}

// fused BN (training-mode batch stats) + tanh, in place
__global__ void bn_tanh_kernel(float* __restrict__ OUT,
                               const float* __restrict__ MU,
                               const float* __restrict__ ISTD,
                               const float* __restrict__ g,
                               const float* __restrict__ bt, int total) {
  int idx = blockIdx.x * 256 + threadIdx.x;
  if (idx >= total) return;
  int c = idx & 127;
  float v = OUT[idx];
  OUT[idx] = tanhf((v - MU[c]) * ISTD[c] * g[c] + bt[c]);
}

// ---------------------------------------------------------------------------
// Launch
// ---------------------------------------------------------------------------
extern "C" void kernel_launch(void* const* d_in, const int* in_sizes, int n_in,
                              void* d_out, int out_size, void* d_ws, size_t ws_size,
                              hipStream_t stream) {
  (void)n_in; (void)out_size; (void)ws_size;
  const int N = 30000, FIN = 256, FOUT = 128;
  const int E = in_sizes[1] / 2;

  const float* x = (const float*)d_in[0];
  const int* eis[3] = {(const int*)d_in[1], (const int*)d_in[2], (const int*)d_in[3]};
  const float* in_g = (const float*)d_in[4];
  const float* in_b = (const float*)d_in[5];

  // workspace carve (256B aligned), ~155 MB total
  char* ws = (char*)d_ws;
  size_t off = 0;
  auto take = [&](size_t bytes) -> void* {
    void* p = ws + off;
    off += (bytes + 255) & ~(size_t)255;
    return p;
  };
  unsigned short* XN    = (unsigned short*)take((size_t)N * FIN * 2);
  float*          SUM   = (float*)take((size_t)N * FIN * 4);
  float*          SSQ   = (float*)take((size_t)N * FIN * 4);
  unsigned*       MAXE  = (unsigned*)take((size_t)N * FIN * 4);
  float*          DEG   = (float*)take((size_t)N * 4);
  unsigned short* STATS = (unsigned short*)take((size_t)N * 768 * 2);
  float*          AMP   = (float*)take((size_t)N * 4);
  unsigned short* WTPRE = (unsigned short*)take((size_t)256 * 512 * 2);
  unsigned short* WQT   = (unsigned short*)take((size_t)128 * 1024 * 2);
  unsigned short* WPT   = (unsigned short*)take((size_t)128 * 768 * 2);
  float*          BCOMB = (float*)take(128 * 4);
  float*          MU_I  = (float*)take(256 * 4);
  float*          IS_I  = (float*)take(256 * 4);
  float*          MU_O  = (float*)take(128 * 4);
  float*          IS_O  = (float*)take(128 * 4);

  // input batchnorm -> bf16
  col_stats_kernel<<<FIN, 256, 0, stream>>>(x, N, FIN, MU_I, IS_I);
  normalize_cast_kernel<<<N, 256, 0, stream>>>(x, MU_I, IS_I, in_g, in_b, XN);

  for (int b = 0; b < 3; ++b) {
    const float* preW  = (const float*)d_in[6 + 8 * b + 0];
    const float* preB  = (const float*)d_in[6 + 8 * b + 1];
    const float* postW = (const float*)d_in[6 + 8 * b + 2];
    const float* postB = (const float*)d_in[6 + 8 * b + 3];
    const float* linW  = (const float*)d_in[6 + 8 * b + 4];
    const float* linB  = (const float*)d_in[6 + 8 * b + 5];
    const float* bnG   = (const float*)d_in[6 + 8 * b + 6];
    const float* bnB   = (const float*)d_in[6 + 8 * b + 7];
    const int* ei = eis[b];
    float* OUT = (float*)d_out + (size_t)b * N * FOUT;

    transpose_preW_kernel<<<(256 * 512) / 256, 256, 0, stream>>>(preW, WTPRE);
    comb_weights_kernel<<<(1792 * 128) / 256, 256, 0, stream>>>(postW, linW, WQT, WPT);
    bcomb_kernel<<<1, 128, 0, stream>>>(postB, linW, linB, BCOMB);

    init_agg_kernel<<<N, 256, 0, stream>>>(SUM, SSQ, MAXE, DEG, N);
    deg_kernel<<<(E + 255) / 256, 256, 0, stream>>>(ei, E, DEG);

    int fullE = E / 32;
    if (fullE)
      edge_gemm_kernel<<<fullE, 128, 0, stream>>>(XN, WTPRE, preB, ei, E, SUM, SSQ, MAXE);
    int tailE = E - fullE * 32;
    if (tailE)
      edge_tail_kernel<<<(tailE * 256 + 255) / 256, 256, 0, stream>>>(
          XN, WTPRE, preB, ei, E, fullE * 32, SUM, SSQ, MAXE);

    finalize_kernel<<<N, 256, 0, stream>>>(SUM, SSQ, MAXE, DEG, STATS, AMP);

    int fullN = N / 16;
    if (fullN)
      node_gemm_kernel<<<fullN, 128, 0, stream>>>(XN, STATS, WQT, WPT, BCOMB, AMP, OUT);
    int tailN = N - fullN * 16;
    if (tailN)
      node_tail_kernel<<<(tailN * 128 + 255) / 256, 256, 0, stream>>>(
          XN, STATS, WQT, WPT, BCOMB, AMP, OUT, fullN * 16, N);

    col_stats_kernel<<<FOUT, 256, 0, stream>>>(OUT, N, FOUT, MU_O, IS_O);
    bn_tanh_kernel<<<(N * FOUT + 255) / 256, 256, 0, stream>>>(OUT, MU_O, IS_O, bnG, bnB,
                                                               N * FOUT);
  }
}
